// PhotonicMamba_59931973648884
// MI455X (gfx1250) — matrix operand order repaired and checked by reading, not compiled, not run on hardware
//
#include <hip/hip_runtime.h>

// ---------------------------------------------------------------- types
typedef __attribute__((ext_vector_type(16))) __bf16 v16bf;
typedef __attribute__((ext_vector_type(4)))  __bf16 bf16x4;
typedef __attribute__((ext_vector_type(8)))  float  v8f;
typedef __attribute__((ext_vector_type(4)))  float  f32x4;
typedef __attribute__((ext_vector_type(4)))  unsigned int u32x4;
typedef __attribute__((ext_vector_type(4)))  int i32x4;

union FragBF { v16bf v; u32x4 q[2]; };

#if defined(__AMDGCN__) && __has_builtin(__builtin_amdgcn_global_load_async_to_lds_b128)
#define PM_ASYNC_LDS 1
typedef __attribute__((address_space(3))) i32x4 lds_i32x4;
#endif

// ---------------------------------------------------------------- sizes
#define DMODEL 1024
#define DINNER 2048
#define DSTATE 16
#define DTRANK 64
#define DCONV  4
#define NBATCH 2
#define SEQLEN 2048
#define MROWS  (NBATCH * SEQLEN)   // 4096

// ================================================================ fp32 -> bf16 (with zero row padding)
__global__ __launch_bounds__(256) void pm_cvt_bf16(const float* __restrict__ src,
                                                   __bf16* __restrict__ dst,
                                                   int n_src, int n_tot) {
    int i = blockIdx.x * 256 + threadIdx.x;
    if (i < n_tot) dst[i] = (i < n_src) ? (__bf16)src[i] : (__bf16)0.f;
}

// ================================================================ generic GEMM
// C[M,N] = A[M,K](f32,row-major,lda) * Bw[Npad,K](bf16,row-major,zero-padded)^T
// mode 0: C = acc     mode 1: C = softplus(acc + bias[n])
// Block: 256 threads (8 waves), 128x128 tile; wave: 32x64 = 2x4 WMMA tiles.
__global__ __launch_bounds__(256) void pm_gemm_bf16(
    const float* __restrict__ A, int lda,
    const __bf16* __restrict__ Bw,
    float* __restrict__ C, int ldc,
    int N, int K,
    const float* __restrict__ bias, int mode) {

    constexpr int BK = 32, LDT = 40;                 // 40 bf16 row pad (20 banks)
    __shared__ __bf16 sA[128 * LDT];                 // 10 KB x2
    __shared__ __bf16 sB[128 * LDT];

    const int tid  = threadIdx.x;
    const int lane = tid & 31;
    const int wid  = tid >> 5;
    const int row0 = blockIdx.y * 128;
    const int col0 = blockIdx.x * 128;
    const int wm   = (wid >> 1) * 32;                // 0,32,64,96
    const int wn   = (wid & 1) * 64;                 // 0,64

    // staging coordinates (fixed per thread)
    const int ar0 = tid >> 3, ac4 = tid & 7;         // A: base row / float4-col
    const int br0 = tid >> 2, bc8 = tid & 3;         // B: base row / uint4-col

    v8f acc[2][4];
    {
        v8f z = {};
        #pragma unroll
        for (int i = 0; i < 2; ++i)
            #pragma unroll
            for (int j = 0; j < 4; ++j) acc[i][j] = z;
    }

    for (int k0 = 0; k0 < K; k0 += BK) {
        // ---- phase 1: issue all global loads (overlap with prior tile's WMMAs)
        f32x4 areg[4];
        #pragma unroll
        for (int i = 0; i < 4; ++i) {
            int r = ar0 + i * 32;
            areg[i] = *(const f32x4*)&A[(size_t)(row0 + r) * lda + (k0 + ac4 * 4)];
        }
        u32x4 breg[2];
        #pragma unroll
        for (int i = 0; i < 2; ++i) {
            int r = br0 + i * 64;
            breg[i] = *(const u32x4*)&Bw[(size_t)(col0 + r) * K + (k0 + bc8 * 8)];
        }
        if (k0 + BK < K) {                           // uniform branch, next-tile prefetch
            __builtin_prefetch(&A[(size_t)(row0 + ar0) * lda + (k0 + BK + ac4 * 4)], 0, 1);
            __builtin_prefetch(&Bw[(size_t)(col0 + br0) * K + (k0 + BK + bc8 * 8)], 0, 1);
        }

        __syncthreads();                             // prior tile's LDS reads done
        // ---- phase 2: convert + store to LDS (vector stores, no branches)
        #pragma unroll
        for (int i = 0; i < 4; ++i) {
            int r = ar0 + i * 32;
            bf16x4 h = { (__bf16)areg[i].x, (__bf16)areg[i].y,
                         (__bf16)areg[i].z, (__bf16)areg[i].w };
            *(bf16x4*)&sA[r * LDT + ac4 * 4] = h;
        }
        #pragma unroll
        for (int i = 0; i < 2; ++i) {
            int r = br0 + i * 64;
            *(u32x4*)&sB[r * LDT + bc8 * 8] = breg[i];
        }
        __syncthreads();

        // ---- phase 3: fragment gather per ISA 16-bit layouts
        // A lane<16: K {0..7,16..23}; lane>=16: K {8..15,24..31}
        FragBF a[2], bfr[4];
        {
            const int arow = wm + (lane & 15);
            const int ac   = (lane < 16) ? 0 : 1;    // uint4 index (0 or 8 bf16)
            #pragma unroll
            for (int i = 0; i < 2; ++i) {
                const u32x4* p = (const u32x4*)&sA[(arow + 16 * i) * LDT];
                a[i].q[0] = p[ac];  a[i].q[1] = p[ac + 2];
            }
        }
        // B lane<16: K {0..7,16..23}; lane>=16: K {8..15,24..31} (column of B = row of sB)
        {
            const int bcol = lane & 15;
            const int bc   = (lane < 16) ? 0 : 1;    // uint4 index (0 or 8 bf16)
            #pragma unroll
            for (int j = 0; j < 4; ++j) {
                const u32x4* p = (const u32x4*)&sB[(wn + 16 * j + bcol) * LDT];
                bfr[j].q[0] = p[bc];  bfr[j].q[1] = p[bc + 2];
            }
        }

        #pragma unroll
        for (int i = 0; i < 2; ++i)
            #pragma unroll
            for (int j = 0; j < 4; ++j)
                acc[i][j] = __builtin_amdgcn_wmma_f32_16x16x32_bf16(
                    false, a[i].v, false, bfr[j].v, (short)0, acc[i][j], false, false);
    }

    // ---- epilogue: D layout VGPR r -> M=r (lanes 0-15) / M=8+r (lanes 16-31), N=lane%16
    const int mofs = (lane < 16) ? 0 : 8;
    const int ncol = lane & 15;
    #pragma unroll
    for (int tm = 0; tm < 2; ++tm) {
        #pragma unroll
        for (int tn = 0; tn < 4; ++tn) {
            int col = col0 + wn + tn * 16 + ncol;
            if (col < N) {
                int rbase = row0 + wm + tm * 16 + mofs;
                #pragma unroll
                for (int r = 0; r < 8; ++r) {
                    float v = acc[tm][tn][r];
                    if (mode == 1) {
                        float x = v + bias[col];
                        v = (x > 20.f) ? x : log1pf(__expf(x));   // softplus
                    }
                    C[(size_t)(rbase + r) * ldc + col] = v;
                }
            }
        }
    }
}

// ================================================================ causal conv1d + silu (4 chans/thread)
__global__ __launch_bounds__(256) void pm_conv_silu(
    const float* __restrict__ xz, const float* __restrict__ cw,
    const float* __restrict__ cb, float* __restrict__ xconv) {
    int idx = blockIdx.x * 256 + threadIdx.x;          // over M * DINNER/4
    if (idx >= MROWS * (DINNER / 4)) return;
    int d4 = (idx << 2) & (DINNER - 1);
    int bt = idx >> 9;                                 // DINNER/4 = 512
    int t  = bt & (SEQLEN - 1);

    f32x4 w[4];
    #pragma unroll
    for (int j = 0; j < 4; ++j) w[j] = *(const f32x4*)&cw[(d4 + j) * DCONV];
    f32x4 acc = *(const f32x4*)&cb[d4];
    #pragma unroll
    for (int k = 0; k < DCONV; ++k) {
        int tt = t - (DCONV - 1) + k;
        if (tt >= 0) {
            f32x4 xv = *(const f32x4*)&xz[(size_t)(bt + tt - t) * (2 * DINNER) + d4];
            acc.x += xv.x * w[0][k];
            acc.y += xv.y * w[1][k];
            acc.z += xv.z * w[2][k];
            acc.w += xv.w * w[3][k];
        }
    }
    f32x4 o;
    o.x = acc.x / (1.f + __expf(-acc.x));
    o.y = acc.y / (1.f + __expf(-acc.y));
    o.z = acc.z / (1.f + __expf(-acc.z));
    o.w = acc.w / (1.f + __expf(-acc.w));
    *(f32x4*)&xconv[(size_t)bt * DINNER + d4] = o;
}

// ================================================================ selective scan (fused gate)
// One lane per (batch, channel); 16-state recurrence in registers.
// B/C staged per 128-step chunk into LDS via ASYNC load-to-LDS when available.
// y overwrites delta in place. Software-pipelined t+1 loads.
__global__ __launch_bounds__(256) void pm_scan(
    const float* __restrict__ xdbl,   // [M,96]: dt|B|C
    float*       delta_y,             // [M,DINNER] in: delta, out: gated y
    const float* __restrict__ xconv,  // [M,DINNER] = u
    const float* __restrict__ xz,     // [M,2*DINNER] (z = cols DINNER..)
    const float* __restrict__ A_log,  // [DINNER,16]
    const float* __restrict__ Dskip)  // [DINNER]
{
    constexpr int TCH = 128;
    __shared__ float sBC[TCH * 32];                // [trel][B0..15,C0..15]

    const int b = blockIdx.x >> 3;                 // 8 blocks of 256 chans per batch
    const int d = ((blockIdx.x & 7) << 8) + threadIdx.x;
    const size_t base = (size_t)b * SEQLEN;

    float Av[DSTATE], h[DSTATE];
    #pragma unroll
    for (int n = 0; n < DSTATE; ++n) {
        Av[n] = -__expf(A_log[d * DSTATE + n]);
        h[n]  = 0.f;
    }
    const float Dv = Dskip[d];

    for (int t0 = 0; t0 < SEQLEN; t0 += TCH) {
        __syncthreads();                           // prior chunk's LDS reads done
        #pragma unroll
        for (int i = 0; i < 4; ++i) {              // stage B/C chunk (1024 float4)
            int e4 = threadIdx.x + i * 256;
            int tr = e4 >> 3, j4 = e4 & 7;
            const float* gp = &xdbl[(base + t0 + tr) * 96 + 64 + j4 * 4];
#ifdef PM_ASYNC_LDS
            // prototype: (i32x4* src_global/generic, i32x4 as(3)* dst_lds, imm off, imm cpol)
            __builtin_amdgcn_global_load_async_to_lds_b128(
                (i32x4*)gp,
                (lds_i32x4*)(unsigned int)(uintptr_t)&sBC[e4 * 4],
                0, 0);
#else
            *(f32x4*)&sBC[e4 * 4] = *(const f32x4*)gp;
#endif
        }
#ifdef PM_ASYNC_LDS
#if __has_builtin(__builtin_amdgcn_s_wait_asynccnt)
        __builtin_amdgcn_s_wait_asynccnt(0);
#else
        asm volatile("s_wait_asynccnt 0" ::: "memory");
#endif
#endif
        __syncthreads();

        size_t m0 = base + t0;
        float dlt = delta_y[m0 * DINNER + d];
        float u   = xconv[m0 * DINNER + d];
        float zv  = xz[m0 * (2 * DINNER) + DINNER + d];

        for (int tr = 0; tr < TCH; ++tr) {
            // issue next step's loads first (hide latency behind exp chain)
            const int trn = (tr + 1 < TCH) ? tr + 1 : tr;
            const size_t mn = base + t0 + trn;
            float ndlt = delta_y[mn * DINNER + d];
            float nu   = xconv[mn * DINNER + d];
            float nzv  = xz[mn * (2 * DINNER) + DINNER + d];

            const float du = dlt * u;
            const float* Bp = &sBC[tr * 32];
            float yv = 0.f;
            #pragma unroll
            for (int n = 0; n < DSTATE; ++n) {
                float dA = __expf(dlt * Av[n]);
                h[n] = dA * h[n] + du * Bp[n];
                yv += h[n] * Bp[16 + n];
            }
            yv = (yv + u * Dv) * (zv / (1.f + __expf(-zv)));   // +D skip, *silu(z)
            delta_y[(base + t0 + tr) * DINNER + d] = yv;       // in-place (read before write)

            dlt = ndlt; u = nu; zv = nzv;
        }
    }
}

// ================================================================ launch
extern "C" void kernel_launch(void* const* d_in, const int* in_sizes, int n_in,
                              void* d_out, int out_size, void* d_ws, size_t ws_size,
                              hipStream_t stream) {
    (void)in_sizes; (void)n_in; (void)out_size; (void)ws_size;

    const float* hidden  = (const float*)d_in[0];  // [B,L,1024]
    const float* W_in    = (const float*)d_in[1];  // [4096,1024]
    const float* conv_w  = (const float*)d_in[2];  // [2048,4]
    const float* conv_b  = (const float*)d_in[3];  // [2048]
    const float* W_xproj = (const float*)d_in[4];  // [96,2048]
    const float* W_dt    = (const float*)d_in[5];  // [2048,64]
    const float* b_dt    = (const float*)d_in[6];  // [2048]
    const float* A_log   = (const float*)d_in[7];  // [2048,16]
    const float* D_skip  = (const float*)d_in[8];  // [2048]
    const float* W_out   = (const float*)d_in[9];  // [1024,2048]
    float* out = (float*)d_out;                    // [B,L,1024]

    // ---- workspace layout (bytes)
    char* ws = (char*)d_ws;
    __bf16* Wb_in  = (__bf16*)(ws + 0);            //  8,388,608  [4096,1024]
    __bf16* Wb_xp  = (__bf16*)(ws + 8388608);      //    524,288  [128(pad),2048]
    __bf16* Wb_dt  = (__bf16*)(ws + 8912896);      //    262,144  [2048,64]
    __bf16* Wb_out = (__bf16*)(ws + 9175040);      //  4,194,304  [1024,2048]
    float*  xz     = (float*)(ws + 13369344);      // 67,108,864
    float*  xconv  = (float*)(ws + 80478208);      // 33,554,432
    float*  xdbl   = (float*)(ws + 114032640);     //  1,572,864
    float*  dly    = (float*)(ws + 115605504);     // 33,554,432 (delta -> y)

    // ---- weight conversion (fp32 -> bf16, layout [N,K], N zero-padded to mult of 128)
    pm_cvt_bf16<<<(2*DINNER*DMODEL + 255)/256, 256, 0, stream>>>(
        W_in, Wb_in, 2*DINNER*DMODEL, 2*DINNER*DMODEL);
    pm_cvt_bf16<<<(128*DINNER + 255)/256, 256, 0, stream>>>(
        W_xproj, Wb_xp, (DTRANK + 2*DSTATE)*DINNER, 128*DINNER);
    pm_cvt_bf16<<<(DINNER*DTRANK + 255)/256, 256, 0, stream>>>(
        W_dt, Wb_dt, DINNER*DTRANK, DINNER*DTRANK);
    pm_cvt_bf16<<<(DMODEL*DINNER + 255)/256, 256, 0, stream>>>(
        W_out, Wb_out, DMODEL*DINNER, DMODEL*DINNER);

    // ---- in_proj: xz[M,4096] = hidden[M,1024] * W_in^T
    pm_gemm_bf16<<<dim3((2*DINNER)/128, MROWS/128), 256, 0, stream>>>(
        hidden, DMODEL, Wb_in, xz, 2*DINNER, 2*DINNER, DMODEL, nullptr, 0);

    // ---- depthwise causal conv + silu
    pm_conv_silu<<<(MROWS*(DINNER/4) + 255)/256, 256, 0, stream>>>(xz, conv_w, conv_b, xconv);

    // ---- x_proj: xdbl[M,96] = xconv[M,2048] * W_xproj^T   (N=96, padded weights)
    pm_gemm_bf16<<<dim3(1, MROWS/128), 256, 0, stream>>>(
        xconv, DINNER, Wb_xp, xdbl, 96, 96, DINNER, nullptr, 0);

    // ---- dt_proj + softplus: delta[M,2048] = softplus(xdbl[:, :64] * W_dt^T + b_dt)
    pm_gemm_bf16<<<dim3(DINNER/128, MROWS/128), 256, 0, stream>>>(
        xdbl, 96, Wb_dt, dly, DINNER, DINNER, DTRANK, b_dt, 1);

    // ---- selective scan + D-skip + silu(z) gate (y written over delta)
    pm_scan<<<NBATCH * (DINNER/256), 256, 0, stream>>>(
        xdbl, dly, xconv, xz, A_log, D_skip);

    // ---- out_proj: out[M,1024] = y[M,2048] * W_out^T
    pm_gemm_bf16<<<dim3(DMODEL/128, MROWS/128), 256, 0, stream>>>(
        dly, DINNER, Wb_out, out, DMODEL, DMODEL, DINNER, nullptr, 0);
}
